// Denoiser_53145925321448
// MI455X (gfx1250) — matrix operand order repaired and checked by reading, not compiled, not run on hardware
//
#include <hip/hip_runtime.h>
#include <hip/hip_bf16.h>

typedef __attribute__((ext_vector_type(16))) _Float16 v16h;
typedef __attribute__((ext_vector_type(8)))  float    v8f;
typedef __attribute__((ext_vector_type(4)))  float    v4f;

static constexpr int NN   = 20000;   // nodes
static constexpr int EE   = 640000;  // edges
static constexpr int DIN  = 92;
static constexpr int DINP = 96;      // padded to multiple of 32
static constexpr int CC   = 128;
static constexpr int HH   = 2;
static constexpr int HC   = HH * CC; // 256
static constexpr int ETOT = EE + NN; // edges + self loops

#define BM 64
#define BN 64

// ---------------------------------------------------------------------------
// Weight pre-pack: f32 [K,M] row-major -> f16 WMMA B-fragments.
// Layout: Bpk[((mt*Kt)+kt)*512 + lane*16 + j]
//   per CDNA5 16-bit B 32x16 layout: lanes 0-15 col=lane K={0..7,16..23};
//   lanes 16-31 col=lane-16 K={8..15,24..31}  (matches A fragment order)
// Zero-pads K up to Kt*32.
// ---------------------------------------------------------------------------
__global__ void pack_weight(const float* __restrict__ Bsrc, _Float16* __restrict__ Bpk,
                            int K, int M, int Kt)
{
    int Mt = M >> 4;
    size_t total = (size_t)Mt * Kt * 512;
    for (size_t i = (size_t)blockIdx.x * blockDim.x + threadIdx.x; i < total;
         i += (size_t)gridDim.x * blockDim.x) {
        int    j    = (int)(i & 15);
        int    lane = (int)((i >> 4) & 31);
        size_t blk  = i >> 9;
        int    kt   = (int)(blk % Kt);
        int    mt   = (int)(blk / Kt);
        int    kb   = (lane < 16) ? 0 : 8;
        int    k    = kt * 32 + kb + (j & 7) + ((j >> 3) << 4);
        int    m    = mt * 16 + (lane & 15);
        float  v    = (k < K) ? Bsrc[(size_t)k * M + m] : 0.0f;
        Bpk[i] = (_Float16)v;
    }
}

// Pad node features [N,92] -> [N,96] (zeros in the tail)
__global__ void pad_feat(const float* __restrict__ src, float* __restrict__ dst, size_t total)
{
    size_t i = (size_t)blockIdx.x * blockDim.x + threadIdx.x;
    if (i < total) {
        int c = (int)(i % DINP);
        size_t r = i / DINP;
        dst[i] = (c < DIN) ? src[r * DIN + c] : 0.0f;
    }
}

// ---------------------------------------------------------------------------
// LDS-free WMMA GEMM: Out[N,M] = A[N,K] @ W (+ bias), K % 32 == 0.
// A: f32 row-major (L2 resident). W: pre-packed f16 fragments (L2 resident).
// Block = 64x64 output, 128 threads (4 waves); wave = 16x64 strip = 4 tiles.
// Per K-step/wave: 4x float4 A loads + 4x v16h B loads + 4 WMMAs.
// ---------------------------------------------------------------------------
__global__ __launch_bounds__(128)
void gemm_wmma_packed(const float* __restrict__ A, const _Float16* __restrict__ Bpk,
                      const float* __restrict__ bias, float* __restrict__ Out,
                      int Nrows, int K, int M)
{
    const int lane    = threadIdx.x & 31;
    const int wave    = threadIdx.x >> 5;
    const int rowBase = blockIdx.x * BM;
    const int colBase = blockIdx.y * BN;
    const int Kt      = K >> 5;
    const int mtBase  = colBase >> 4;

    // A fragment addressing (CDNA5 16-bit A 16x32 layout):
    // lanes 0-15: row=lane,    K = {kb..kb+7, kb+16..kb+23}, kb=0
    // lanes16-31: row=lane-16, same pattern with kb=8
    const int  row   = rowBase + wave * 16 + (lane & 15);
    const bool rowOk = row < Nrows;
    const int  aKb   = (lane < 16) ? 0 : 8;
    const float* Arow = A + (size_t)row * K;

    v8f acc[4];
    #pragma unroll
    for (int t = 0; t < 4; ++t)
        #pragma unroll
        for (int j = 0; j < 8; ++j) acc[t][j] = 0.0f;

    for (int kt = 0; kt < Kt; ++kt) {
        const int kb = kt * 32 + aKb;
        v4f f0, f1, f2, f3;
        if (rowOk) {
            f0 = *(const v4f*)(Arow + kb);
            f1 = *(const v4f*)(Arow + kb + 4);
            f2 = *(const v4f*)(Arow + kb + 16);
            f3 = *(const v4f*)(Arow + kb + 20);
        } else {
            f0 = f1 = f2 = f3 = (v4f){0.f, 0.f, 0.f, 0.f};
        }
        v16h afrag;
        #pragma unroll
        for (int j = 0; j < 4; ++j) {
            afrag[j]      = (_Float16)f0[j];
            afrag[4 + j]  = (_Float16)f1[j];
            afrag[8 + j]  = (_Float16)f2[j];
            afrag[12 + j] = (_Float16)f3[j];
        }

        const _Float16* bp = Bpk + (((size_t)mtBase * Kt + kt) << 9) + lane * 16;
        #pragma unroll
        for (int t = 0; t < 4; ++t) {
            const v16h bfrag = *(const v16h*)(bp + ((size_t)t * Kt << 9));
            acc[t] = __builtin_amdgcn_wmma_f32_16x16x32_f16(
                false, afrag, false, bfrag, (short)0, acc[t], false, false);
        }
    }

    // C/D layout: VGPR v, lanes 0-15 -> M=v, lanes 16-31 -> M=v+8; N = lane&15
    const int rOff = (lane >> 4) * 8;
    #pragma unroll
    for (int t = 0; t < 4; ++t) {
        #pragma unroll
        for (int v = 0; v < 8; ++v) {
            int gr = rowBase + wave * 16 + rOff + v;
            int gc = colBase + t * 16 + (lane & 15);
            if (gr < Nrows) {
                float o = acc[t][v];
                if (bias) o += bias[gc];
                Out[(size_t)gr * M + gc] = o;
            }
        }
    }
}

// ---------------------------------------------------------------------------
// BatchNorm (training mode) helpers
// ---------------------------------------------------------------------------
__global__ __launch_bounds__(256)
void bn_stats(const float* __restrict__ X, int Nrows, int C,
              float* __restrict__ mu, float* __restrict__ rsig)
{
    int c = blockIdx.x;
    __shared__ float ssum[256], ssq[256];
    float s = 0.0f, q = 0.0f;
    for (int r = threadIdx.x; r < Nrows; r += 256) {
        float v = X[(size_t)r * C + c];
        s += v; q += v * v;
    }
    ssum[threadIdx.x] = s; ssq[threadIdx.x] = q;
    __syncthreads();
    for (int off = 128; off > 0; off >>= 1) {
        if (threadIdx.x < off) {
            ssum[threadIdx.x] += ssum[threadIdx.x + off];
            ssq[threadIdx.x]  += ssq[threadIdx.x + off];
        }
        __syncthreads();
    }
    if (threadIdx.x == 0) {
        float m   = ssum[0] / (float)Nrows;
        float var = ssq[0] / (float)Nrows - m * m;
        mu[c]   = m;
        rsig[c] = rsqrtf(var + 1e-5f);
    }
}

__global__ __launch_bounds__(256)
void bn_apply_relu(float* __restrict__ X, const float* __restrict__ mu,
                   const float* __restrict__ rsig, const float* __restrict__ g,
                   const float* __restrict__ be, size_t total, int C)
{
    size_t i = (size_t)blockIdx.x * blockDim.x + threadIdx.x;
    if (i < total) {
        int c = (int)(i % C);
        float v = (X[i] - mu[c]) * rsig[c] * g[c] + be[c];
        X[i] = v > 0.0f ? v : 0.0f;
    }
}

// ---------------------------------------------------------------------------
// Elementwise / fill utilities
// ---------------------------------------------------------------------------
__global__ void fill_f32(float* p, float v, size_t n) {
    size_t i = (size_t)blockIdx.x * blockDim.x + threadIdx.x;
    if (i < n) p[i] = v;
}
__global__ void fill_u32(unsigned* p, unsigned v, size_t n) {
    size_t i = (size_t)blockIdx.x * blockDim.x + threadIdx.x;
    if (i < n) p[i] = v;
}
__global__ void add_inplace_k(float* __restrict__ dst, const float* __restrict__ a, size_t n) {
    size_t i = (size_t)blockIdx.x * blockDim.x + threadIdx.x;
    if (i < n) dst[i] += a[i];
}
__global__ void copy_k(float* __restrict__ dst, const float* __restrict__ s, size_t n) {
    size_t i = (size_t)blockIdx.x * blockDim.x + threadIdx.x;
    if (i < n) dst[i] = s[i];
}
__global__ void bias_row_add_k(float* __restrict__ X, const float* __restrict__ b,
                               size_t total, int M) {
    size_t i = (size_t)blockIdx.x * blockDim.x + threadIdx.x;
    if (i < total) X[i] += b[i % M];
}

// ---------------------------------------------------------------------------
// GIN: agg[dst] += x[src]  (one block handles 2 edges x 128 channels)
// ---------------------------------------------------------------------------
__global__ __launch_bounds__(256)
void gin_gather(const int* __restrict__ ei, const float* __restrict__ X,
                float* __restrict__ Agg, int E_)
{
    int e = blockIdx.x * 2 + (threadIdx.x >> 7);
    int c = threadIdx.x & 127;
    if (e < E_) {
        int s = ei[e];
        int d = ei[E_ + e];
        atomicAdd(&Agg[(size_t)d * CC + c], X[(size_t)s * CC + c]);
    }
}

// ---------------------------------------------------------------------------
// VAE reparameterization: z = eps*exp(0.5*logvar)+mu  (also writes zin output)
// ---------------------------------------------------------------------------
__global__ void vae_reparam(const float* __restrict__ eps, const float* __restrict__ mu,
                            const float* __restrict__ logvar, float* __restrict__ z,
                            float* __restrict__ zin, size_t total)
{
    size_t i = (size_t)blockIdx.x * blockDim.x + threadIdx.x;
    if (i < total) {
        float v = eps[i] * __expf(0.5f * logvar[i]) + mu[i];
        z[i] = v; zin[i] = v;
    }
}

// ---------------------------------------------------------------------------
// GAT pieces
// ---------------------------------------------------------------------------
__device__ __forceinline__ unsigned fenc(float f) {
    unsigned u = __float_as_uint(f);
    return (u & 0x80000000u) ? ~u : (u | 0x80000000u);
}
__device__ __forceinline__ float fdec(unsigned u) {
    u = (u & 0x80000000u) ? (u & 0x7FFFFFFFu) : ~u;
    return __uint_as_float(u);
}
__device__ __forceinline__ void edge_sd(const int* __restrict__ ei, int E_, int e,
                                        int& s, int& d) {
    if (e < E_) { s = ei[e]; d = ei[E_ + e]; }
    else        { s = d = e - E_; }            // self loop
}

__global__ __launch_bounds__(256)
void gat_attn_coef(const float* __restrict__ hh, const float* __restrict__ att_src,
                   const float* __restrict__ att_dst, float* __restrict__ as_,
                   float* __restrict__ ad_)
{
    int i = blockIdx.x * blockDim.x + threadIdx.x;     // n*H + h
    if (i < NN * HH) {
        int n = i / HH, h = i % HH;
        const float* row = hh + (size_t)n * HC + (size_t)h * CC;
        float s = 0.0f, d = 0.0f;
        for (int c = 0; c < CC; ++c) {
            float v = row[c];
            s += v * att_src[h * CC + c];
            d += v * att_dst[h * CC + c];
        }
        as_[i] = s; ad_[i] = d;
    }
}

__global__ __launch_bounds__(256)
void edge_logit_max(const int* __restrict__ ei, const float* __restrict__ as_,
                    const float* __restrict__ ad_, float* __restrict__ logits,
                    unsigned* __restrict__ maxenc)
{
    int i = blockIdx.x * blockDim.x + threadIdx.x;     // e*H + h
    if (i < ETOT * HH) {
        int e = i / HH, h = i % HH;
        int s, d; edge_sd(ei, EE, e, s, d);
        float lg = as_[s * HH + h] + ad_[d * HH + h];
        lg = lg > 0.0f ? lg : 0.2f * lg;               // leaky_relu(0.2)
        logits[i] = lg;
        atomicMax(&maxenc[d * HH + h], fenc(lg));
    }
}

__global__ __launch_bounds__(256)
void edge_expsum(const int* __restrict__ ei, float* __restrict__ logits,
                 const unsigned* __restrict__ maxenc, float* __restrict__ sum)
{
    int i = blockIdx.x * blockDim.x + threadIdx.x;
    if (i < ETOT * HH) {
        int e = i / HH, h = i % HH;
        int s, d; edge_sd(ei, EE, e, s, d);
        (void)s;
        float ex = __expf(logits[i] - fdec(maxenc[d * HH + h]));
        logits[i] = ex;                                 // in-place: now holds exp
        atomicAdd(&sum[d * HH + h], ex);
    }
}

__global__ __launch_bounds__(256)
void edge_scatter(const int* __restrict__ ei, const float* __restrict__ exv,
                  const float* __restrict__ sum, const float* __restrict__ hh,
                  float* __restrict__ outg)
{
    int e = blockIdx.x;
    int h = threadIdx.x >> 7;                          // /128
    int c = threadIdx.x & 127;
    if (e < ETOT) {
        int s, d; edge_sd(ei, EE, e, s, d);
        float alpha = exv[e * HH + h] / sum[d * HH + h];
        atomicAdd(&outg[(size_t)d * HC + h * CC + c],
                  alpha * hh[(size_t)s * HC + h * CC + c]);
    }
}

// ---------------------------------------------------------------------------
// Host-side orchestration
// ---------------------------------------------------------------------------
static inline void launch_gemm(const float* A, const _Float16* Bpk, const float* bias,
                               float* Out, int n, int k, int m, hipStream_t s)
{
    dim3 grid((n + BM - 1) / BM, (m + BN - 1) / BN);
    gemm_wmma_packed<<<grid, 128, 0, s>>>(A, Bpk, bias, Out, n, k, m);
}
static inline dim3 g1d(size_t n) { return dim3((unsigned)((n + 255) / 256)); }

extern "C" void kernel_launch(void* const* d_in, const int* in_sizes, int n_in,
                              void* d_out, int out_size, void* d_ws, size_t ws_size,
                              hipStream_t stream)
{
    (void)in_sizes; (void)n_in; (void)out_size; (void)ws_size;

    const float* node_feat = (const float*)d_in[0];
    const int*   ei        = (const int*)  d_in[1];   // [2,E]
    const float* eps       = (const float*)d_in[2];
    const float* W_emb = (const float*)d_in[3],  *b_emb = (const float*)d_in[4];
    const float* g_emb = (const float*)d_in[5],  *be_emb= (const float*)d_in[6];
    const float* W1    = (const float*)d_in[7],  *b1    = (const float*)d_in[8];
    const float* g1_   = (const float*)d_in[9],  *be1   = (const float*)d_in[10];
    const float* W2    = (const float*)d_in[11], *b2    = (const float*)d_in[12];
    const float* W_mu  = (const float*)d_in[13], *b_mu  = (const float*)d_in[14];
    const float* W_var = (const float*)d_in[15], *b_var = (const float*)d_in[16];
    const float* W_gat = (const float*)d_in[17];
    const float* att_s = (const float*)d_in[18], *att_d = (const float*)d_in[19];
    const float* b_gat = (const float*)d_in[20];
    const float* W_dec = (const float*)d_in[21], *b_dec = (const float*)d_in[22];

    float* out_zin    = (float*)d_out;
    float* out_zout   = (float*)d_out + (size_t)NN * CC;
    float* out_mu     = (float*)d_out + (size_t)2 * NN * CC;
    float* out_logvar = (float*)d_out + (size_t)3 * NN * CC;

    // ---- workspace carve ----
    char* w = (char*)d_ws;
    auto carve = [&](size_t bytes) -> void* {
        void* p = (void*)w;
        w += (bytes + 255) & ~(size_t)255;
        return p;
    };
    const size_t NC  = (size_t)NN * CC;
    const size_t NHC = (size_t)NN * HC;
    float*    bufA    = (float*)   carve(NC  * 4);   // activations x
    float*    bufB    = (float*)   carve(NC  * 4);   // agg / h
    float*    bufC    = (float*)   carve(NC  * 4);   // MLP hidden
    float*    bufZ    = (float*)   carve(NC  * 4);   // z / zout
    float*    hh      = (float*)   carve(NHC * 4);   // [N,H,C]
    float*    outg    = (float*)   carve(NHC * 4);   // GAT aggregation
    float*    featpad = (float*)   carve((size_t)NN * DINP * 4);
    float*    as_     = (float*)   carve((size_t)NN * HH * 4);
    float*    ad_     = (float*)   carve((size_t)NN * HH * 4);
    float*    sumb    = (float*)   carve((size_t)NN * HH * 4);
    unsigned* maxenc  = (unsigned*)carve((size_t)NN * HH * 4);
    float*    logits  = (float*)   carve((size_t)ETOT * HH * 4);
    float*    bn_mu   = (float*)   carve(CC * 4);
    float*    bn_rs   = (float*)   carve(CC * 4);
    // packed f16 weights (fragment order)
    _Float16* pk_emb = (_Float16*)carve((size_t)3 * 8  * 512 * 2);  // K=96,  M=128
    _Float16* pk_W1  = (_Float16*)carve((size_t)4 * 8  * 512 * 2);  // K=128, M=128
    _Float16* pk_W2  = (_Float16*)carve((size_t)4 * 8  * 512 * 2);
    _Float16* pk_mu  = (_Float16*)carve((size_t)4 * 8  * 512 * 2);
    _Float16* pk_var = (_Float16*)carve((size_t)4 * 8  * 512 * 2);
    _Float16* pk_gat = (_Float16*)carve((size_t)4 * 16 * 512 * 2);  // K=128, M=256
    _Float16* pk_dec = (_Float16*)carve((size_t)8 * 8  * 512 * 2);  // K=256, M=128

    // ---- 0. pre-pack weights + pad features ----
    pack_weight<<<64, 256, 0, stream>>>(W_emb, pk_emb, DIN, CC, 3);
    pack_weight<<<64, 256, 0, stream>>>(W1,    pk_W1,  CC,  CC, 4);
    pack_weight<<<64, 256, 0, stream>>>(W2,    pk_W2,  CC,  CC, 4);
    pack_weight<<<64, 256, 0, stream>>>(W_mu,  pk_mu,  CC,  CC, 4);
    pack_weight<<<64, 256, 0, stream>>>(W_var, pk_var, CC,  CC, 4);
    pack_weight<<<64, 256, 0, stream>>>(W_gat, pk_gat, CC,  HC, 4);
    pack_weight<<<64, 256, 0, stream>>>(W_dec, pk_dec, HC,  CC, 8);
    pad_feat<<<g1d((size_t)NN * DINP), 256, 0, stream>>>(node_feat, featpad, (size_t)NN * DINP);

    // ---- 1. atom embedding: x = relu(bn(feat @ W_emb + b_emb)) ----
    launch_gemm(featpad, pk_emb, b_emb, bufA, NN, DINP, CC, stream);
    bn_stats<<<CC, 256, 0, stream>>>(bufA, NN, CC, bn_mu, bn_rs);
    bn_apply_relu<<<g1d(NC), 256, 0, stream>>>(bufA, bn_mu, bn_rs, g_emb, be_emb, NC, CC);

    // ---- 2. GIN encoder, 2 shared-weight layers ----
    for (int l = 0; l < 2; ++l) {
        fill_f32<<<g1d(NC), 256, 0, stream>>>(bufB, 0.0f, NC);
        gin_gather<<<(EE + 1) / 2, 256, 0, stream>>>(ei, bufA, bufB, EE);
        add_inplace_k<<<g1d(NC), 256, 0, stream>>>(bufB, bufA, NC); // h = x + agg
        launch_gemm(bufB, pk_W1, b1, bufC, NN, CC, CC, stream);
        bn_stats<<<CC, 256, 0, stream>>>(bufC, NN, CC, bn_mu, bn_rs);
        bn_apply_relu<<<g1d(NC), 256, 0, stream>>>(bufC, bn_mu, bn_rs, g1_, be1, NC, CC);
        launch_gemm(bufC, pk_W2, b2, bufA, NN, CC, CC, stream);
    }

    // ---- 3. VAE heads + reparameterization ----
    launch_gemm(bufA, pk_mu,  b_mu,  out_mu,     NN, CC, CC, stream);
    launch_gemm(bufA, pk_var, b_var, out_logvar, NN, CC, CC, stream);
    vae_reparam<<<g1d(NC), 256, 0, stream>>>(eps, out_mu, out_logvar, bufZ, out_zin, NC);

    // ---- 4. GAT decoder, 2 shared-weight layers (with self loops) ----
    for (int l = 0; l < 2; ++l) {
        launch_gemm(bufZ, pk_gat, nullptr, hh, NN, CC, HC, stream);
        gat_attn_coef<<<g1d((size_t)NN * HH), 256, 0, stream>>>(hh, att_s, att_d, as_, ad_);
        fill_u32<<<g1d((size_t)NN * HH), 256, 0, stream>>>(maxenc, 0x007FFFFFu, (size_t)NN * HH);
        edge_logit_max<<<g1d((size_t)ETOT * HH), 256, 0, stream>>>(ei, as_, ad_, logits, maxenc);
        fill_f32<<<g1d((size_t)NN * HH), 256, 0, stream>>>(sumb, 0.0f, (size_t)NN * HH);
        edge_expsum<<<g1d((size_t)ETOT * HH), 256, 0, stream>>>(ei, logits, maxenc, sumb);
        fill_f32<<<g1d(NHC), 256, 0, stream>>>(outg, 0.0f, NHC);
        edge_scatter<<<ETOT, 256, 0, stream>>>(ei, logits, sumb, hh, outg);
        bias_row_add_k<<<g1d(NHC), 256, 0, stream>>>(outg, b_gat, NHC, HC);
        launch_gemm(outg, pk_dec, b_dec, bufZ, NN, HC, CC, stream);
    }

    copy_k<<<g1d(NC), 256, 0, stream>>>(out_zout, bufZ, NC);
}
